// EdgeBlock_48034914239037
// MI455X (gfx1250) — matrix operand order repaired and checked by reading, not compiled, not run on hardware
//
#include <hip/hip_runtime.h>

typedef __attribute__((ext_vector_type(16))) __bf16 v16bf;
typedef __attribute__((ext_vector_type(8)))  float  v8f;

#define DD        128
#define IN_DIM    384
#define HIDW      128
#define EPB       128     // edges per block (8 waves * 16)
#define THREADS   256

// LDS partition (in ushort units). Total = 131072 ushorts = 262144 bytes.
#define W1F_OFF   0                    // 12*8*32*16 = 49152
#define W2F_OFF   49152                // 4*8*32*16  = 16384
#define XW_OFF    65536                // 8 waves * 6144
#define HW_OFF    114688               // 8 waves * 2048
#define SMEM_BYTES 262144

__device__ __forceinline__ unsigned short f2bf(float f) {
    unsigned u = __builtin_bit_cast(unsigned, f);
    unsigned r = u + 0x7FFFu + ((u >> 16) & 1u);   // round-to-nearest-even
    return (unsigned short)(r >> 16);
}

extern __shared__ unsigned short smem[];

__global__ void __launch_bounds__(THREADS, 1)
edge_block_wmma(const float* __restrict__ node_attr,
                const float* __restrict__ edge_attr,
                const int*   __restrict__ snd,
                const int*   __restrict__ rcv,
                const float* __restrict__ W1,
                const float* __restrict__ b1,
                const float* __restrict__ W2,
                const float* __restrict__ b2,
                const float* __restrict__ ln_g,
                const float* __restrict__ ln_b,
                float* __restrict__ out,
                int nEdges)
{
    const int tid  = threadIdx.x;
    const int wv   = tid >> 5;
    const int lane = tid & 31;
    const int lo   = lane & 15;
    const int half = lane >> 4;

    unsigned short* w1f = smem + W1F_OFF;
    unsigned short* w2f = smem + W2F_OFF;
    unsigned short* xw  = smem + XW_OFF + wv * 6144;
    unsigned short* hw  = smem + HW_OFF + wv * 2048;

    // ---- Stage W1 (384x128 f32) -> bf16, fragment-linear B layout ----
    // B frag: lane dl holds col N=dl&15, K = chunk*32 + (dl>>4)*16 + e
    for (int idx = tid; idx < IN_DIM * HIDW; idx += THREADS) {
        int K = idx >> 7, N = idx & 127;
        int k0 = K >> 5, kin = K & 31;
        int e  = kin & 15;
        int dl = (kin >> 4) * 16 + (N & 15);
        int n0 = N >> 4;
        w1f[(((k0 * 8 + n0) * 32 + dl) << 4) + e] = f2bf(W1[idx]);
    }
    // ---- Stage W2 (128x128) the same way ----
    for (int idx = tid; idx < HIDW * DD; idx += THREADS) {
        int K = idx >> 7, N = idx & 127;
        int k0 = K >> 5, kin = K & 31;
        int e  = kin & 15;
        int dl = (kin >> 4) * 16 + (N & 15);
        int n0 = N >> 4;
        w2f[(((k0 * 8 + n0) * 32 + dl) << 4) + e] = f2bf(W2[idx]);
    }

    // ---- Stage X tile: gather [sender|receiver|edge] rows -> A-fragment layout ----
    // A frag: lane holds row M=lane&15, K = chunk*32 + (lane>>4)*16 + e
    const int eBase = blockIdx.x * EPB + wv * 16;
    for (int i = lane; i < 1536; i += 32) {          // 16 rows * 96 col-quads
        int row = i / 96;
        int col = (i % 96) * 4;
        int eg  = eBase + row;
        int ec  = (eg < nEdges) ? eg : (nEdges - 1);
        const float* src;
        if (col < 128)      src = node_attr + (size_t)snd[ec] * DD + col;
        else if (col < 256) src = node_attr + (size_t)rcv[ec] * DD + (col - 128);
        else                src = edge_attr + (size_t)ec  * DD + (col - 256);
        float4 v = *(const float4*)src;
        int k0 = col >> 5, kin = col & 31;
        int e  = kin & 15;   // 4-aligned
        int dl = (kin >> 4) * 16 + row;
        int di = ((k0 * 32 + dl) << 4) + e;
        uint2 pk;
        pk.x = (unsigned)f2bf(v.x) | ((unsigned)f2bf(v.y) << 16);
        pk.y = (unsigned)f2bf(v.z) | ((unsigned)f2bf(v.w) << 16);
        *(uint2*)(xw + di) = pk;
    }
    __syncthreads();

    // ---- Layer 1: h = relu(X(16x384) * W1(384x128) + b1), 12 K-chunks x 8 N-tiles ----
    v8f z = {0.f, 0.f, 0.f, 0.f, 0.f, 0.f, 0.f, 0.f};
    v8f acc[8];
#pragma unroll
    for (int n = 0; n < 8; ++n) acc[n] = z;

    for (int k0 = 0; k0 < 12; ++k0) {
        v16bf a = *(const v16bf*)(xw + ((k0 * 32 + lane) << 4));
#pragma unroll
        for (int n = 0; n < 8; ++n) {
            v16bf b = *(const v16bf*)(w1f + (((k0 * 8 + n) * 32 + lane) << 4));
            acc[n] = __builtin_amdgcn_wmma_f32_16x16x32_bf16(
                false, a, false, b, (short)0, acc[n], false, false);
        }
    }

    // ---- bias + ReLU, write H back to LDS in layer-2 A-fragment layout ----
#pragma unroll
    for (int n = 0; n < 8; ++n) {
        float b1n = b1[n * 16 + lo];
#pragma unroll
        for (int r = 0; r < 8; ++r) {
            float h = acc[n][r] + b1n;
            h = h > 0.f ? h : 0.f;
            int M   = r + 8 * half;
            int K   = n * 16 + lo;
            int k0  = K >> 5, kin = K & 31;
            int e   = kin & 15;
            int dl  = (kin >> 4) * 16 + M;
            hw[((k0 * 32 + dl) << 4) + e] = f2bf(h);
        }
    }
    __syncthreads();

    // ---- Layer 2: out = H(16x128) * W2(128x128) + b2 ----
#pragma unroll
    for (int n = 0; n < 8; ++n) acc[n] = z;

    for (int k0 = 0; k0 < 4; ++k0) {
        v16bf a = *(const v16bf*)(hw + ((k0 * 32 + lane) << 4));
#pragma unroll
        for (int n = 0; n < 8; ++n) {
            v16bf b = *(const v16bf*)(w2f + (((k0 * 8 + n) * 32 + lane) << 4));
            acc[n] = __builtin_amdgcn_wmma_f32_16x16x32_bf16(
                false, a, false, b, (short)0, acc[n], false, false);
        }
    }

    // ---- LayerNorm over 128 channels per edge row ----
    float s1[8], s2[8];
#pragma unroll
    for (int r = 0; r < 8; ++r) { s1[r] = 0.f; s2[r] = 0.f; }
#pragma unroll
    for (int n = 0; n < 8; ++n) {
        float b2n = b2[n * 16 + lo];
#pragma unroll
        for (int r = 0; r < 8; ++r) {
            float v = acc[n][r] + b2n;
            acc[n][r] = v;
            s1[r] += v;
            s2[r] += v * v;
        }
    }
    // reduce across the 16 lanes of each half-wave (row lives in one half)
#pragma unroll
    for (int r = 0; r < 8; ++r) {
#pragma unroll
        for (int m = 1; m <= 8; m <<= 1) {
            s1[r] += __shfl_xor(s1[r], m, 32);
            s2[r] += __shfl_xor(s2[r], m, 32);
        }
    }

#pragma unroll
    for (int n = 0; n < 8; ++n) {
        float g  = ln_g[n * 16 + lo];
        float bt = ln_b[n * 16 + lo];
#pragma unroll
        for (int r = 0; r < 8; ++r) {
            float mean = s1[r] * (1.f / 128.f);
            float var  = s2[r] * (1.f / 128.f) - mean * mean;
            float rs   = rsqrtf(var + 1e-5f);
            float o    = (acc[n][r] - mean) * rs * g + bt;
            int M  = r + 8 * half;
            int eg = eBase + M;
            if (eg < nEdges)
                out[(size_t)eg * DD + n * 16 + lo] = o;
        }
    }
}

extern "C" void kernel_launch(void* const* d_in, const int* in_sizes, int n_in,
                              void* d_out, int out_size, void* d_ws, size_t ws_size,
                              hipStream_t stream) {
    (void)n_in; (void)out_size; (void)d_ws; (void)ws_size;
    const float* node_attr = (const float*)d_in[0];
    const float* edge_attr = (const float*)d_in[1];
    const int*   snd       = (const int*)d_in[2];
    const int*   rcv       = (const int*)d_in[3];
    const float* W1        = (const float*)d_in[4];
    const float* b1        = (const float*)d_in[5];
    const float* W2        = (const float*)d_in[6];
    const float* b2        = (const float*)d_in[7];
    const float* ln_g      = (const float*)d_in[8];
    const float* ln_b      = (const float*)d_in[9];
    float* out = (float*)d_out;

    int nEdges = in_sizes[2];
    int blocks = (nEdges + EPB - 1) / EPB;
    if (blocks <= 0) return;

    hipFuncSetAttribute(reinterpret_cast<const void*>(edge_block_wmma),
                        hipFuncAttributeMaxDynamicSharedMemorySize, SMEM_BYTES);
    edge_block_wmma<<<blocks, THREADS, SMEM_BYTES, stream>>>(
        node_attr, edge_attr, snd, rcv, W1, b1, W2, b2, ln_g, ln_b, out, nEdges);
}
